// SAGE_32512902431457
// MI455X (gfx1250) — matrix operand order repaired and checked by reading, not compiled, not run on hardware
//
#include <hip/hip_runtime.h>
#include <hip/hip_bf16.h>

// ---------------------------------------------------------------------------
// GraphSAGE (3 layers, D=128) + center-pair MLP for MI455X (gfx1250).
//  - Features stored bf16 end-to-end; aggregation in f32 via hw atomics.
//  - GEMMs: v_wmma_f32_16x16x32_bf16 (f32 accumulate), weights staged in LDS,
//    A fragments are raw b128 loads (no conversion VALU in hot loop).
// ---------------------------------------------------------------------------

typedef __bf16 bf16_t;
typedef __attribute__((ext_vector_type(16))) __bf16 v16bf;
typedef __attribute__((ext_vector_type(8)))  float  v8f;

#define TD 128            // hidden dim
#define WPITCH (TD + 8)   // LDS row pitch (bf16 elems): +16B breaks bank conflicts

__device__ __forceinline__ __bf16 f2bf(float f) {
  unsigned u = __builtin_bit_cast(unsigned, f);
  u += 0x7FFFu + ((u >> 16) & 1u);                 // round-to-nearest-even
  unsigned short hs = (unsigned short)(u >> 16);
  return __builtin_bit_cast(__bf16, hs);
}
__device__ __forceinline__ unsigned pack2bf(float a, float b) {
  unsigned la = __builtin_bit_cast(unsigned short, f2bf(a));
  unsigned lb = __builtin_bit_cast(unsigned short, f2bf(b));
  return la | (lb << 16);
}
__device__ __forceinline__ float bf2f_lo(unsigned u) { return __builtin_bit_cast(float, u << 16); }
__device__ __forceinline__ float bf2f_hi(unsigned u) { return __builtin_bit_cast(float, u & 0xffff0000u); }

// ------------------------------- utility kernels ---------------------------

__global__ void zero_f32_kernel(float* __restrict__ p, int n) {
  int i = blockIdx.x * blockDim.x + threadIdx.x;
  if (i < n) p[i] = 0.0f;
}

__global__ void zero_f4_kernel(float4* __restrict__ p, int n4) {
  int i = blockIdx.x * blockDim.x + threadIdx.x;
  if (i < n4) p[i] = make_float4(0.f, 0.f, 0.f, 0.f);
}

__global__ void cvt_bf16_kernel(const float* __restrict__ s, bf16_t* __restrict__ d, int n) {
  int i = blockIdx.x * blockDim.x + threadIdx.x;
  if (i < n) d[i] = f2bf(s[i]);
}

// x[n,:] = bf16(z_table[z[n],:])  (4 elems per lane, 32 lanes per node)
__global__ void embed_kernel(const int* __restrict__ z, const float* __restrict__ table,
                             bf16_t* __restrict__ x, int N) {
  int t = blockIdx.x * blockDim.x + threadIdx.x;
  int n = t >> 5, c = (t & 31) * 4;
  if (n < N) {
    int zz = z[n];
    float4 v = *(const float4*)(table + (size_t)zz * TD + c);
    uint2 o;
    o.x = pack2bf(v.x, v.y);
    o.y = pack2bf(v.z, v.w);
    *(uint2*)(x + (size_t)n * TD + c) = o;
  }
}

__global__ void degree_kernel(const int* __restrict__ dst, float* __restrict__ counts, int E) {
  int e = blockIdx.x * blockDim.x + threadIdx.x;
  if (e < E) unsafeAtomicAdd(&counts[dst[e]], 1.0f);
}

__global__ void invdeg_kernel(const float* __restrict__ counts, float* __restrict__ inv, int N) {
  int i = blockIdx.x * blockDim.x + threadIdx.x;
  if (i < N) inv[i] = 1.0f / fmaxf(counts[i], 1.0f);
}

// agg[dst,:] += f32(x[src,:]); one wave per edge, 4 elems/lane, f32 hw atomics.
__global__ void scatter_kernel(const bf16_t* __restrict__ x, const int* __restrict__ src,
                               const int* __restrict__ dst, float* __restrict__ agg, int E) {
  int t = blockIdx.x * blockDim.x + threadIdx.x;
  int e = t >> 5, lane = t & 31;
  if (e >= E) return;
  int s = src[e], d = dst[e];
  uint2 v = *(const uint2*)(x + (size_t)s * TD + lane * 4);
  float* p = agg + (size_t)d * TD + lane * 4;
  unsafeAtomicAdd(p + 0, bf2f_lo(v.x));
  unsafeAtomicAdd(p + 1, bf2f_hi(v.x));
  unsafeAtomicAdd(p + 2, bf2f_lo(v.y));
  unsafeAtomicAdd(p + 3, bf2f_hi(v.y));
}

// aggb[n,:] = bf16(agg[n,:] * inv_deg[n])  -- folds mean + conversion
__global__ void scalecvt_kernel(const float* __restrict__ agg, const float* __restrict__ inv,
                                bf16_t* __restrict__ aggb, int N) {
  int t = blockIdx.x * blockDim.x + threadIdx.x;
  int n = t >> 5, c = (t & 31) * 4;
  if (n >= N) return;
  float s = inv[n];
  float4 v = *(const float4*)(agg + (size_t)n * TD + c);
  uint2 o;
  o.x = pack2bf(v.x * s, v.y * s);
  o.y = pack2bf(v.z * s, v.w * s);
  *(uint2*)(aggb + (size_t)n * TD + c) = o;
}

// -------------------------- WMMA GEMM (the hot path) -----------------------
// Y = bf16( relu?( A0 @ W0^T  [+ A1 @ W1b^T]  + bias ) )
// Block = 4 waves; weights staged in LDS once per block; one wave computes a
// 16-row x 128-col strip. K = 128 = 4 steps of k32; 8/16 WMMAs per step.
template <bool HAS_B>
__global__ __launch_bounds__(128)
void gemm_bf16_kernel(const bf16_t* __restrict__ A0, const bf16_t* __restrict__ W0,
                      const bf16_t* __restrict__ A1, const bf16_t* __restrict__ W1b,
                      const float* __restrict__ bias,
                      bf16_t* __restrict__ Y, int M, int relu) {
  constexpr int NMAT = HAS_B ? 2 : 1;
  __shared__ bf16_t smW[NMAT][TD][WPITCH];

  // cooperative weight staging: 16B (8 bf16) chunks, 2048 chunks per matrix
  for (int idx = threadIdx.x; idx < TD * (TD / 8); idx += 128) {
    const int row = idx >> 4;       // 16 chunks per row
    const int ch  = (idx & 15) * 8;
    *(uint4*)&smW[0][row][ch] = *(const uint4*)(W0 + (size_t)row * TD + ch);
    if constexpr (HAS_B)
      *(uint4*)&smW[1][row][ch] = *(const uint4*)(W1b + (size_t)row * TD + ch);
  }
  __syncthreads();

  const int lane = threadIdx.x & 31;
  const int wave = threadIdx.x >> 5;
  const int m0 = (blockIdx.x * 4 + wave) * 16;
  if (m0 >= M) return;                 // wave-uniform: EXEC stays all-ones
  const int r = lane & 15;             // A row-in-tile / B-C column-in-tile
  const int h = lane >> 4;             // lane-half selector

  int rowA = m0 + r;                   // clamp for partial last tile (loads only)
  if (rowA > M - 1) rowA = M - 1;

  v8f acc[8];
#pragma unroll
  for (int t = 0; t < 8; t++) acc[t] = (v8f)0.0f;

  const bf16_t* pa0 = A0 + (size_t)rowA * TD;
  const bf16_t* pa1 = HAS_B ? (A1 + (size_t)rowA * TD) : A0;

#pragma unroll
  for (int kb = 0; kb < TD; kb += 32) {
    // A fragment (16x32 bf16): elems 0..7 = K kb+8h+0..7 ; 8..15 = K kb+16+8h+0..7
    union { v16bf v; uint4 q[2]; } a0, a1;
    a0.q[0] = *(const uint4*)(pa0 + kb + 8 * h);
    a0.q[1] = *(const uint4*)(pa0 + kb + 16 + 8 * h);
    if constexpr (HAS_B) {
      a1.q[0] = *(const uint4*)(pa1 + kb + 8 * h);
      a1.q[1] = *(const uint4*)(pa1 + kb + 16 + 8 * h);
    }
#pragma unroll
    for (int t = 0; t < 8; t++) {
      // B fragment (32x16 bf16): lane col n = t*16+r, elems 0..7 = W[n][kb+8h+0..7] ; 8..15 = W[n][kb+16+8h+0..7]
      const int wrow = t * 16 + r;
      union { v16bf v; uint4 q[2]; } b0;
      b0.q[0] = *(const uint4*)&smW[0][wrow][kb + 8 * h];
      b0.q[1] = *(const uint4*)&smW[0][wrow][kb + 16 + 8 * h];
      acc[t] = __builtin_amdgcn_wmma_f32_16x16x32_bf16(false, a0.v, false, b0.v,
                                                       (short)0, acc[t], false, false);
      if constexpr (HAS_B) {
        union { v16bf v; uint4 q[2]; } b1;
        b1.q[0] = *(const uint4*)&smW[1][wrow][kb + 8 * h];
        b1.q[1] = *(const uint4*)&smW[1][wrow][kb + 16 + 8 * h];
        acc[t] = __builtin_amdgcn_wmma_f32_16x16x32_bf16(false, a1.v, false, b1.v,
                                                         (short)0, acc[t], false, false);
      }
    }
  }

  // Epilogue: C/D layout -> m = m0 + vr + 8h, n = t*16 + r
#pragma unroll
  for (int t = 0; t < 8; t++) {
    const int col = t * 16 + r;
    const float bv = bias ? bias[col] : 0.0f;
#pragma unroll
    for (int vr = 0; vr < 8; vr++) {
      const int m = m0 + vr + 8 * h;
      if (m < M) {
        float v = acc[t][vr] + bv;
        if (relu) v = fmaxf(v, 0.0f);
        Y[(size_t)m * TD + col] = f2bf(v);
      }
    }
  }
}

// ------------------------------ readout kernels ----------------------------

// centers[g] = lower_bound(batch, g)  (batch is sorted)
__global__ void centers_kernel(const int* __restrict__ batch, int* __restrict__ centers,
                               int N, int G) {
  int g = blockIdx.x * blockDim.x + threadIdx.x;
  if (g >= G) return;
  int lo = 0, hi = N;
  while (lo < hi) { int mid = (lo + hi) >> 1; if (batch[mid] < g) lo = mid + 1; else hi = mid; }
  centers[g] = lo;
}

// hin[g,:] = x[centers[g],:] * x[centers[g]+1,:]   (bf16 in/out, f32 math)
__global__ void pair_kernel(const bf16_t* __restrict__ x, const int* __restrict__ centers,
                            bf16_t* __restrict__ hin, int G) {
  int t = blockIdx.x * blockDim.x + threadIdx.x;
  int g = t >> 5, c = (t & 31) * 4;
  if (g >= G) return;
  int c0 = centers[g];
  uint2 ra = *(const uint2*)(x + (size_t)c0 * TD + c);
  uint2 rb = *(const uint2*)(x + (size_t)(c0 + 1) * TD + c);
  uint2 o;
  o.x = pack2bf(bf2f_lo(ra.x) * bf2f_lo(rb.x), bf2f_hi(ra.x) * bf2f_hi(rb.x));
  o.y = pack2bf(bf2f_lo(ra.y) * bf2f_lo(rb.y), bf2f_hi(ra.y) * bf2f_hi(rb.y));
  *(uint2*)(hin + (size_t)g * TD + c) = o;
}

// out[g] = dot(h[g,:], W2[0,:]) + b2   (one wave per row, xor-shuffle reduce)
__global__ void out_kernel(const bf16_t* __restrict__ h, const float* __restrict__ W2,
                           const float* __restrict__ b2, float* __restrict__ out, int G) {
  int t = blockIdx.x * blockDim.x + threadIdx.x;
  int g = t >> 5, lane = t & 31;
  if (g >= G) return;
  uint2 hv = *(const uint2*)(h + (size_t)g * TD + lane * 4);
  float4 wv = *(const float4*)(W2 + lane * 4);
  float s = bf2f_lo(hv.x) * wv.x + bf2f_hi(hv.x) * wv.y +
            bf2f_lo(hv.y) * wv.z + bf2f_hi(hv.y) * wv.w;
#pragma unroll
  for (int off = 16; off > 0; off >>= 1) s += __shfl_xor(s, off, 32);
  if (lane == 0) out[g] = s + b2[0];
}

// ------------------------------- launcher ----------------------------------

extern "C" void kernel_launch(void* const* d_in, const int* in_sizes, int n_in,
                              void* d_out, int out_size, void* d_ws, size_t ws_size,
                              hipStream_t stream) {
  const int N = in_sizes[0];
  const int E = in_sizes[1] / 2;
  const int G = out_size;
  const int L = in_sizes[4] / (TD * TD);

  const int*   z       = (const int*)d_in[0];
  const int*   src     = (const int*)d_in[1];
  const int*   dst     = src + E;
  const int*   batch   = (const int*)d_in[2];
  const float* z_table = (const float*)d_in[3];
  const float* Wl      = (const float*)d_in[4];
  const float* bl      = (const float*)d_in[5];
  const float* Wr      = (const float*)d_in[6];
  const float* W1      = (const float*)d_in[7];
  const float* b1      = (const float*)d_in[8];
  const float* W2      = (const float*)d_in[9];
  const float* b2      = (const float*)d_in[10];
  float*       out     = (float*)d_out;

  // carve scratch
  char* ws = (char*)d_ws;
  auto carve = [&](size_t bytes) -> char* {
    char* p = ws; ws += (bytes + 255) & ~(size_t)255; return p;
  };
  float*  agg     = (float*)carve((size_t)N * TD * 4);   // f32 (atomics)
  bf16_t* xA      = (bf16_t*)carve((size_t)N * TD * 2);
  bf16_t* xB      = (bf16_t*)carve((size_t)N * TD * 2);
  bf16_t* aggb    = (bf16_t*)carve((size_t)N * TD * 2);
  float*  counts  = (float*)carve((size_t)N * 4);
  float*  inv     = (float*)carve((size_t)N * 4);
  int*    centers = (int*)carve((size_t)G * 4);
  bf16_t* hin     = (bf16_t*)carve((size_t)G * TD * 2);
  bf16_t* hbuf    = (bf16_t*)carve((size_t)G * TD * 2);
  bf16_t* Wl_bf   = (bf16_t*)carve((size_t)L * TD * TD * 2);
  bf16_t* Wr_bf   = (bf16_t*)carve((size_t)L * TD * TD * 2);
  bf16_t* W1_bf   = (bf16_t*)carve((size_t)TD * TD * 2);

  const int B = 256;
  auto nb = [&](long n) { return (unsigned)((n + B - 1) / B); };

  // weights -> bf16 (once per call; deterministic)
  const int nW = L * TD * TD;
  cvt_bf16_kernel<<<nb(nW), B, 0, stream>>>(Wl, Wl_bf, nW);
  cvt_bf16_kernel<<<nb(nW), B, 0, stream>>>(Wr, Wr_bf, nW);
  cvt_bf16_kernel<<<nb(TD * TD), B, 0, stream>>>(W1, W1_bf, TD * TD);

  // degrees + embedding
  zero_f32_kernel<<<nb(N), B, 0, stream>>>(counts, N);
  degree_kernel<<<nb(E), B, 0, stream>>>(dst, counts, E);
  invdeg_kernel<<<nb(N), B, 0, stream>>>(counts, inv, N);
  embed_kernel<<<nb((long)N * 32), B, 0, stream>>>(z, z_table, xA, N);

  // SAGE layers
  bf16_t* xcur = xA;
  bf16_t* xnext = xB;
  for (int l = 0; l < L; l++) {
    zero_f4_kernel<<<nb((long)N * 32), B, 0, stream>>>((float4*)agg, N * 32);
    scatter_kernel<<<nb((long)E * 32), B, 0, stream>>>(xcur, src, dst, agg, E);
    scalecvt_kernel<<<nb((long)N * 32), B, 0, stream>>>(agg, inv, aggb, N);
    const int mtiles = (N + 15) / 16;
    gemm_bf16_kernel<true><<<(unsigned)((mtiles + 3) / 4), 128, 0, stream>>>(
        aggb, Wl_bf + (size_t)l * TD * TD,
        xcur, Wr_bf + (size_t)l * TD * TD,
        bl + (size_t)l * TD, xnext, N, (l < L - 1) ? 1 : 0);
    bf16_t* t = xcur; xcur = xnext; xnext = t;
  }

  // readout MLP
  centers_kernel<<<nb(G), B, 0, stream>>>(batch, centers, N, G);
  pair_kernel<<<nb((long)G * 32), B, 0, stream>>>(xcur, centers, hin, G);
  {
    const int mtiles = (G + 15) / 16;
    gemm_bf16_kernel<false><<<(unsigned)((mtiles + 3) / 4), 128, 0, stream>>>(
        hin, W1_bf, nullptr, nullptr, b1, hbuf, G, 1);
  }
  out_kernel<<<nb((long)G * 32), B, 0, stream>>>(hbuf, W2, b2, out, G);
}
